// SelfAttention_18940805776016
// MI455X (gfx1250) — matrix operand order repaired and checked by reading, not compiled, not run on hardware
//
#include <hip/hip_runtime.h>
#include <hip/hip_bf16.h>

// ---------------------------------------------------------------------------
// Self-attention forward for MI455X (gfx1250).
//   bf16 WMMA (f32 accumulate); A and B panels staged into LDS by the
//   Tensor Data Mover (tensor_load_to_lds) with pad_enable bank-skewing.
//   B=4, S=2048, D=1024.
//   d_out = [ out (4*2048*1024 f32) | attention (4*2048*2048 f32) ]
// ---------------------------------------------------------------------------

typedef __bf16 bf16_t;
typedef __attribute__((ext_vector_type(16))) __bf16 v16bf;
typedef __attribute__((ext_vector_type(8)))  __bf16 v8bf;
typedef __attribute__((ext_vector_type(4)))  __bf16 v4bf;
typedef __attribute__((ext_vector_type(8)))  float  v8f;
typedef __attribute__((ext_vector_type(4)))  float  v4f;
typedef unsigned int u32x4 __attribute__((ext_vector_type(4)));
typedef int          i32x8 __attribute__((ext_vector_type(8)));
typedef int          i32x4 __attribute__((ext_vector_type(4)));

#define B_SZ 4
#define S_SZ 2048
#define D_SZ 1024

// LDS panel geometry: 128 rows x 64 k-elements (bf16), rows padded to 72
// elements (144 B = 36 DWORDs). 36n mod 64 is distinct for n=0..15, so the 16
// fragment lanes hit distinct bank groups; 144 B keeps 16 B alignment.
#define PANEL_ROWS 128
#define PANEL_K    64
#define ROW_PITCH  72

#if defined(__HIP_DEVICE_COMPILE__) && __has_builtin(__builtin_amdgcn_tensor_load_to_lds)
#define USE_TDM 1
#else
#define USE_TDM 0
#endif

// ---------------------------------------------------------------------------
__device__ __forceinline__ v8f wmma_bf16(v16bf a, v16bf b, v8f c) {
#if defined(__HIP_DEVICE_COMPILE__)
    return __builtin_amdgcn_wmma_f32_16x16x32_bf16(
        false, a, false, b, (short)0, c, false, false);
#else
    (void)a; (void)b;
    return c;
#endif
}

__device__ __forceinline__ void wait_tensorcnt0() {
#if defined(__HIP_DEVICE_COMPILE__)
#if __has_builtin(__builtin_amdgcn_s_wait_tensorcnt)
    __builtin_amdgcn_s_wait_tensorcnt(0);
#else
    asm volatile("s_wait_tensorcnt 0x0" ::: "memory");
#endif
#endif
}

// ---------------------------------------------------------------------------
// TDM 2-D tile load Global -> LDS, bf16, tile = PANEL_ROWS x PANEL_K.
// Descriptor per CDNA5 ISA §8.3/8.4. g1[0]:
//   data_size=1 (2B) | pad_enable | pad_interval=4 (32 DW = one 128 B row)
//   | pad_amount=3 (4 DW = 16 B)  => LDS row pitch 144 B.
// ---------------------------------------------------------------------------
#if USE_TDM
__device__ __forceinline__ void tdm_load_panel(
    unsigned ldsAddr, const bf16_t* gsrc,
    unsigned tensorK, unsigned tensorRows, unsigned rowStride)
{
    const unsigned long long ga = (unsigned long long)(size_t)gsrc;
    u32x4 g0;
    g0.x = 1u;                                          // count=1
    g0.y = ldsAddr;                                     // lds_addr
    g0.z = (unsigned)(ga & 0xffffffffu);                // global_addr lo
    g0.w = (unsigned)((ga >> 32) & 0x01ffffffu) | (2u << 30); // hi | type=2

    i32x8 g1;
    g1[0] = (int)0x07110000u;                           // 2B | pad 16B/128B
    g1[1] = (int)((tensorK & 0xffffu) << 16);           // tensor_dim0 lo
    g1[2] = (int)(((tensorK >> 16) & 0xffffu) |
                  ((tensorRows & 0xffffu) << 16));      // td0 hi | td1 lo
    g1[3] = (int)(((tensorRows >> 16) & 0xffffu) |
                  ((unsigned)PANEL_K << 16));           // td1 hi | tile_dim0
    g1[4] = (int)(PANEL_ROWS & 0xffffu);                // tile_dim1, tile_dim2=0
    g1[5] = (int)rowStride;                             // tensor_dim0_stride lo
    g1[6] = 0;
    g1[7] = 0;

    i32x4 gz = {0, 0, 0, 0};
#if defined(__clang_major__) && (__clang_major__ >= 23)
    i32x8 gz8 = {0, 0, 0, 0, 0, 0, 0, 0};
    __builtin_amdgcn_tensor_load_to_lds(g0, g1, gz, gz, gz8, 0);
#else
    __builtin_amdgcn_tensor_load_to_lds(g0, g1, gz, gz, 0);
#endif
}
#else
// Cooperative fallback: same padded layout (ROW_PITCH elements per row).
__device__ __forceinline__ void coop_copy_panel(
    bf16_t* __restrict__ dst, const bf16_t* __restrict__ src, int ld, int tid)
{
#pragma unroll
    for (int c = 0; c < 4; ++c) {
        const int chunk = tid + c * 256;        // 0..1023 (128 rows x 8 chunks)
        const int row   = chunk >> 3;
        const int off   = (chunk & 7) * 8;      // 8 bf16 = 16 B
        *(v8bf*)(dst + row * ROW_PITCH + off) =
            *(const v8bf*)(src + (size_t)row * ld + off);
    }
}
#endif

// ---------------------------------------------------------------------------
// LDS fragment loaders (ISA 7.12.2 wave32 layouts, padded rows)
// A 16x32: lane l -> m=l&15, kh=l>>4; K = ksub+kh*8+[0..8) and +16
// B 32x16: lane l -> n=l&15, kh=l>>4; K = ksub+kh*8+[0..8) and +16
// ---------------------------------------------------------------------------
__device__ __forceinline__ v16bf lds_frag_a(const bf16_t* rowBase, int ksub, int lane) {
    const int m  = lane & 15;
    const int kh = lane >> 4;
    const bf16_t* p = rowBase + m * ROW_PITCH + ksub + kh * 8;
    v8bf lo = *(const v8bf*)p;
    v8bf hi = *(const v8bf*)(p + 16);
    v16bf r;
#pragma unroll
    for (int i = 0; i < 8; ++i) { r[i] = lo[i]; r[i + 8] = hi[i]; }
    return r;
}

__device__ __forceinline__ v16bf lds_frag_b(const bf16_t* rowBase, int ksub, int lane) {
    const int n  = lane & 15;
    const int kh = lane >> 4;
    const bf16_t* p = rowBase + n * ROW_PITCH + ksub + kh * 8;
    v8bf lo = *(const v8bf*)p;
    v8bf hi = *(const v8bf*)(p + 16);
    v16bf r;
#pragma unroll
    for (int i = 0; i < 8; ++i) { r[i] = lo[i]; r[i + 8] = hi[i]; }
    return r;
}

// ---------------------------------------------------------------------------
// GEMM:  D[m][n] = scale * sum_k A[m][k] * Bmat[n][k]  (+ bias[n])
// Block: 8 waves as 4(M) x 2(N); block tile 128M x 128N; K-stage = 64.
// Wave 0 issues two TDM panel loads per stage (A rows, B rows), double
// buffered; all operands come from LDS. Per k-substep, all fragments are
// preloaded into registers so the 8 WMMAs issue as a dense pack behind a
// single dscnt wait.
// ---------------------------------------------------------------------------
template <bool OUT_BF16, bool TRANS>
__global__ __launch_bounds__(256) void gemm_wmma(
    const bf16_t* __restrict__ Amat, const bf16_t* __restrict__ Bmat,
    const float* __restrict__ bias, void* __restrict__ Dv,
    int M, int N, int K, int lda, int ldb, int ldd, float scale,
    long long sA, long long sB, long long sD)
{
    __shared__ bf16_t ldsA[2][PANEL_ROWS][ROW_PITCH];   // 36 KiB
    __shared__ bf16_t ldsB[2][PANEL_ROWS][ROW_PITCH];   // 36 KiB

    const int lane   = threadIdx.x & 31;
    const int waveId = threadIdx.x >> 5;
    const int mw = waveId >> 1;            // 0..3 : M sub-block
    const int nw = waveId & 1;             // 0..1 : N half

    const int nBlocks = N >> 7;
    const int mBlock  = blockIdx.x / nBlocks;
    const int nBlock  = blockIdx.x - mBlock * nBlocks;
    const int blockM0 = mBlock << 7;
    const int blockN0 = nBlock << 7;
    const long long z = blockIdx.z;

    const bf16_t* Ab = Amat + z * sA + (size_t)blockM0 * lda;
    const bf16_t* Bb = Bmat + z * sB + (size_t)blockN0 * ldb;

    v8f acc[8] = {};

    // ---- prologue: stage k-stage 0 into buffer 0 ----
#if USE_TDM
    if (waveId == 0) {
        tdm_load_panel((unsigned)(size_t)&ldsA[0][0][0], Ab,
                       (unsigned)K, (unsigned)M, (unsigned)lda);
        tdm_load_panel((unsigned)(size_t)&ldsB[0][0][0], Bb,
                       (unsigned)K, (unsigned)N, (unsigned)ldb);
        wait_tensorcnt0();
    }
#else
    coop_copy_panel(&ldsA[0][0][0], Ab, lda, threadIdx.x);
    coop_copy_panel(&ldsB[0][0][0], Bb, ldb, threadIdx.x);
#endif
    __syncthreads();

    int cur = 0;
    for (int kk = 0; kk < K; kk += PANEL_K) {
        const int kk2 = kk + PANEL_K;
        const int nxt = cur ^ 1;

        // --- async-stage next panels while computing current ---
        if (kk2 < K) {
#if USE_TDM
            if (waveId == 0) {
                tdm_load_panel((unsigned)(size_t)&ldsA[nxt][0][0], Ab + kk2,
                               (unsigned)K, (unsigned)M, (unsigned)lda);
                tdm_load_panel((unsigned)(size_t)&ldsB[nxt][0][0], Bb + kk2,
                               (unsigned)K, (unsigned)N, (unsigned)ldb);
            }
#else
            coop_copy_panel(&ldsA[nxt][0][0], Ab + kk2, lda, threadIdx.x);
            coop_copy_panel(&ldsB[nxt][0][0], Bb + kk2, ldb, threadIdx.x);
#endif
        }

        // --- 2 k-substeps; per substep: preload all fragments, then a dense
        //     pack of 8 WMMAs (2 M-subtiles x 4 N-subtiles) ---
#pragma unroll
        for (int ks = 0; ks < 2; ++ks) {
            const int ksub = ks * 32;
            v16bf a0 = lds_frag_a(&ldsA[cur][mw * 32][0],      ksub, lane);
            v16bf a1 = lds_frag_a(&ldsA[cur][mw * 32 + 16][0], ksub, lane);
            v16bf bf[4];
#pragma unroll
            for (int j = 0; j < 4; ++j)
                bf[j] = lds_frag_b(&ldsB[cur][nw * 64 + j * 16][0], ksub, lane);
#pragma unroll
            for (int j = 0; j < 4; ++j) {
                acc[j]     = wmma_bf16(a0, bf[j], acc[j]);
                acc[4 + j] = wmma_bf16(a1, bf[j], acc[4 + j]);
            }
        }

#if USE_TDM
        if (waveId == 0 && kk2 < K) wait_tensorcnt0();
#endif
        __syncthreads();
        cur = nxt;
    }

    // ---- epilogue: lane owns column n; VGPR r -> row r + 8*(lane>>4) ----
    const int n  = lane & 15;
    const int mh = lane >> 4;
    const int rowA0 = blockM0 + mw * 32;
#pragma unroll
    for (int j = 0; j < 4; ++j) {
        const int ng = blockN0 + nw * 64 + j * 16 + n;
        const float bv = bias ? bias[ng] : 0.0f;
#pragma unroll
        for (int ms = 0; ms < 2; ++ms) {
#pragma unroll
            for (int r = 0; r < 8; ++r) {
                const int mg = rowA0 + ms * 16 + r + (mh << 3);
                const float v = acc[ms * 4 + j][r] * scale + bv;
                if (OUT_BF16) {
                    bf16_t* Dp = (bf16_t*)Dv + z * sD;
                    if (TRANS) Dp[(size_t)ng * ldd + mg] = (bf16_t)v;
                    else       Dp[(size_t)mg * ldd + ng] = (bf16_t)v;
                } else {
                    float* Dp = (float*)Dv + z * sD;
                    Dp[(size_t)mg * ldd + ng] = v;
                }
            }
        }
    }
}

// ---------------------------------------------------------------------------
// f32 -> bf16 bulk convert (4 elems/thread)
// ---------------------------------------------------------------------------
__global__ __launch_bounds__(256) void cvt_f32_bf16(
    const float* __restrict__ src, bf16_t* __restrict__ dst, long long n4)
{
    const long long i = (long long)blockIdx.x * blockDim.x + threadIdx.x;
    if (i >= n4) return;
    v4f v = *(const v4f*)(src + i * 4);
    v4bf o;
#pragma unroll
    for (int k = 0; k < 4; ++k) o[k] = (bf16_t)v[k];
    *(v4bf*)(dst + i * 4) = o;
}

// ---------------------------------------------------------------------------
// In-place row softmax over S f32 elements; one block per row.
// Also writes a bf16 copy of the probabilities for the P@V WMMA GEMM.
// ---------------------------------------------------------------------------
__global__ __launch_bounds__(256) void softmax_rows(
    float* __restrict__ attn, bf16_t* __restrict__ probsBf, int S)
{
    __shared__ float red[256];
    const long long row = blockIdx.x;
    float*  p  = attn    + row * (long long)S;
    bf16_t* pb = probsBf + row * (long long)S;
    const int t = threadIdx.x;

    float mx = -3.402823466e+38f;
    for (int i = t; i < S; i += 256) mx = fmaxf(mx, p[i]);
    red[t] = mx;
    __syncthreads();
    for (int s = 128; s > 0; s >>= 1) {
        if (t < s) red[t] = fmaxf(red[t], red[t + s]);
        __syncthreads();
    }
    mx = red[0];
    __syncthreads();

    float sum = 0.0f;
    for (int i = t; i < S; i += 256) {
        float e = __expf(p[i] - mx);
        p[i] = e;
        sum += e;
    }
    red[t] = sum;
    __syncthreads();
    for (int s = 128; s > 0; s >>= 1) {
        if (t < s) red[t] += red[t + s];
        __syncthreads();
    }
    const float inv = 1.0f / red[0];
    for (int i = t; i < S; i += 256) {
        const float v = p[i] * inv;
        p[i]  = v;
        pb[i] = (bf16_t)v;
    }
}

// ---------------------------------------------------------------------------
// Host launcher
// ---------------------------------------------------------------------------
extern "C" void kernel_launch(void* const* d_in, const int* in_sizes, int n_in,
                              void* d_out, int out_size, void* d_ws, size_t ws_size,
                              hipStream_t stream)
{
    (void)in_sizes; (void)n_in; (void)out_size; (void)ws_size;

    const float* x  = (const float*)d_in[0];
    const float* Wk = (const float*)d_in[1];
    const float* bk = (const float*)d_in[2];
    const float* Wq = (const float*)d_in[3];
    const float* bq = (const float*)d_in[4];
    const float* Wv = (const float*)d_in[5];
    const float* bv = (const float*)d_in[6];
    const float* Wo = (const float*)d_in[7];
    const float* bo = (const float*)d_in[8];

    float* out  = (float*)d_out;                         // [4,2048,1024]
    float* attn = out + (size_t)B_SZ * S_SZ * D_SZ;      // [4,2048,2048]

    // Workspace layout (bf16 staging), 120 MiB total
    char* ws = (char*)d_ws;
    bf16_t* xb  = (bf16_t*)(ws + (0LL  << 20));  // 16 MiB  x bf16 [8192,1024]
    bf16_t* WkB = (bf16_t*)(ws + (16LL << 20));
    bf16_t* WqB = (bf16_t*)(ws + (18LL << 20));
    bf16_t* WvB = (bf16_t*)(ws + (20LL << 20));
    bf16_t* WoB = (bf16_t*)(ws + (22LL << 20));
    bf16_t* Qb  = (bf16_t*)(ws + (24LL << 20));  // 16 MiB  [4][2048][1024]
    bf16_t* Kb  = (bf16_t*)(ws + (40LL << 20));  // 16 MiB
    bf16_t* Vt  = (bf16_t*)(ws + (56LL << 20));  // 16 MiB  V^T: [4][1024][2048]
    bf16_t* Ob  = (bf16_t*)(ws + (72LL << 20));  // 16 MiB  attn-out bf16
    bf16_t* Pb  = (bf16_t*)(ws + (88LL << 20));  // 32 MiB  probs bf16 [4][2048][2048]

    const long long MD = (long long)B_SZ * S_SZ * D_SZ;
    const long long DD = (long long)D_SZ * D_SZ;
    const long long SD = (long long)S_SZ * D_SZ;
    const long long SS = (long long)S_SZ * S_SZ;

    // ---- convert inputs to bf16 ----
    cvt_f32_bf16<<<dim3((unsigned)(MD / 4 / 256)), 256, 0, stream>>>(x, xb, MD / 4);
    cvt_f32_bf16<<<dim3((unsigned)(DD / 4 / 256)), 256, 0, stream>>>(Wk, WkB, DD / 4);
    cvt_f32_bf16<<<dim3((unsigned)(DD / 4 / 256)), 256, 0, stream>>>(Wq, WqB, DD / 4);
    cvt_f32_bf16<<<dim3((unsigned)(DD / 4 / 256)), 256, 0, stream>>>(Wv, WvB, DD / 4);
    cvt_f32_bf16<<<dim3((unsigned)(DD / 4 / 256)), 256, 0, stream>>>(Wo, WoB, DD / 4);

    // ---- Q = x@Wq^T+bq ; K = x@Wk^T+bk ----
    gemm_wmma<true, false><<<dim3(512, 1, 1), 256, 0, stream>>>(
        xb, WqB, bq, Qb, 8192, 1024, 1024, 1024, 1024, 1024, 1.0f, 0, 0, 0);
    gemm_wmma<true, false><<<dim3(512, 1, 1), 256, 0, stream>>>(
        xb, WkB, bk, Kb, 8192, 1024, 1024, 1024, 1024, 1024, 1.0f, 0, 0, 0);

    // ---- V^T = (x@Wv^T+bv)^T stored [b][d][s] ----
    gemm_wmma<true, true><<<dim3(128, 1, B_SZ), 256, 0, stream>>>(
        xb, WvB, bv, Vt, 2048, 1024, 1024, 1024, 1024, 2048, 1.0f,
        SD, 0, (long long)D_SZ * S_SZ);

    // ---- scores = (Q@K^T) * 1/sqrt(D), f32 into attention region ----
    gemm_wmma<false, false><<<dim3(256, 1, B_SZ), 256, 0, stream>>>(
        Qb, Kb, (const float*)nullptr, attn, 2048, 2048, 1024, 1024, 1024, 2048,
        0.03125f, SD, SD, SS);

    // ---- softmax rows, in place; emit bf16 probs for P@V ----
    softmax_rows<<<dim3(B_SZ * S_SZ), 256, 0, stream>>>(attn, Pb, S_SZ);

    // ---- attn_out = P @ V ----
    gemm_wmma<true, false><<<dim3(128, 1, B_SZ), 256, 0, stream>>>(
        Pb, Vt, (const float*)nullptr, Ob, 2048, 1024, 2048,
        2048, 2048, 1024, 1.0f, SS, (long long)D_SZ * S_SZ, SD);

    // ---- out = attn_out @ Wo^T + bo ----
    gemm_wmma<false, false><<<dim3(512, 1, 1), 256, 0, stream>>>(
        Ob, WoB, bo, out, 8192, 1024, 1024, 1024, 1024, 1024, 1.0f, 0, 0, 0);
}
